// Net_54090818125832
// MI455X (gfx1250) — matrix operand order repaired and checked by reading, not compiled, not run on hardware
//
#include <hip/hip_runtime.h>
#include <hip/hip_bf16.h>

typedef __attribute__((ext_vector_type(16))) _Float16 v16h;
typedef __attribute__((ext_vector_type(8)))  float    v8f;

#define BB   64
#define SS   512
#define HH   768
#define NT   36           // NTAG
#define TSTART 34
#define TSTOP  35
#define KT   24           // 768 / 32 k-tiles
#define NEG_INF (-1e30f)

__device__ __forceinline__ float lrelu(float x) { return x >= 0.f ? x : 0.01f * x; }

// ---------------------------------------------------------------------------
// Kernel 1: xlen[b] = max_s head[b,s]
// ---------------------------------------------------------------------------
__global__ __launch_bounds__(256) void xlen_kernel(const int* __restrict__ hidx,
                                                   int* __restrict__ xlen) {
    __shared__ int sm[256];
    int b = blockIdx.x, t = threadIdx.x;
    int m = hidx[b * SS + t];
    int m2 = hidx[b * SS + 256 + t];
    sm[t] = m > m2 ? m : m2;
    __syncthreads();
    for (int s = 128; s > 0; s >>= 1) {
        if (t < s) sm[t] = sm[t] > sm[t + s] ? sm[t] : sm[t + s];
        __syncthreads();
    }
    if (t == 0) xlen[b] = sm[0];
}

// ---------------------------------------------------------------------------
// Kernel 2: pre-pack W (H x 36, f32) into WMMA B-fragments (f16, N padded to 48)
// fragment id = jt*KT + kt ; per lane 16 contiguous halfs.
// B 32x16 layout: lane -> N = lane&15, K = 8*(lane>>4)+v (v<8), 16+8*(lane>>4)+(v-8) (v>=8).
// ---------------------------------------------------------------------------
__global__ __launch_bounds__(32) void packw_kernel(const float* __restrict__ W,
                                                   _Float16* __restrict__ wfrag) {
    int frag = blockIdx.x;          // 0..71
    int lane = threadIdx.x;         // 0..31
    int jt = frag / KT, kt = frag % KT;
    int col  = jt * 16 + (lane & 15);
    int kbase = kt * 32 + ((lane >> 4) << 3);
    _Float16* out = wfrag + ((size_t)frag * 32 + lane) * 16;
    for (int v = 0; v < 16; ++v) {
        float val = (col < NT) ? W[(kbase + v + (v >= 8 ? 8 : 0)) * NT + col] : 0.f;
        out[v] = (_Float16)val;
    }
}

// ---------------------------------------------------------------------------
// Kernel 3: gathered GEMM + bias + leaky_relu via v_wmma_f32_16x16x32_f16.
// 8 waves / block, 1 M-tile (16 rows of B*S) per wave, 3 accumulators (N=0..47).
// ---------------------------------------------------------------------------
__global__ __launch_bounds__(256) void gemm_logits_kernel(
        const float* __restrict__ x1, const int* __restrict__ hidx,
        const float* __restrict__ bias, const _Float16* __restrict__ wfrag,
        float* __restrict__ logits) {
    const int wave = threadIdx.x >> 5;
    const int lane = threadIdx.x & 31;
    const int mtile = blockIdx.x * 8 + wave;        // 2048 tiles
    const int m    = lane & 15;
    const int row  = mtile * 16 + m;                // flat (b,s) row
    const int bidx = row >> 9;
    const int s    = row & (SS - 1);
    const int hi   = hidx[bidx * SS + s];
    const float* arow = x1 + ((size_t)bidx * SS + hi) * HH;
    const int off = (lane >> 4) * 8;                // A-layout K sub-offset

    v8f acc0 = {}, acc1 = {}, acc2 = {};
    for (int kt = 0; kt < KT; ++kt) {
        const int k0 = kt * 32;
        const float4* p0 = reinterpret_cast<const float4*>(arow + k0 + off);
        const float4* p1 = reinterpret_cast<const float4*>(arow + k0 + 16 + off);
        float4 f0 = p0[0], f1 = p0[1], f2 = p1[0], f3 = p1[1];
        v16h a;
        a[0] = (_Float16)f0.x;  a[1] = (_Float16)f0.y;
        a[2] = (_Float16)f0.z;  a[3] = (_Float16)f0.w;
        a[4] = (_Float16)f1.x;  a[5] = (_Float16)f1.y;
        a[6] = (_Float16)f1.z;  a[7] = (_Float16)f1.w;
        a[8] = (_Float16)f2.x;  a[9] = (_Float16)f2.y;
        a[10] = (_Float16)f2.z; a[11] = (_Float16)f2.w;
        a[12] = (_Float16)f3.x; a[13] = (_Float16)f3.y;
        a[14] = (_Float16)f3.z; a[15] = (_Float16)f3.w;

        v16h b0 = *reinterpret_cast<const v16h*>(wfrag + ((size_t)(0 * KT + kt) * 32 + lane) * 16);
        v16h b1 = *reinterpret_cast<const v16h*>(wfrag + ((size_t)(1 * KT + kt) * 32 + lane) * 16);
        v16h b2 = *reinterpret_cast<const v16h*>(wfrag + ((size_t)(2 * KT + kt) * 32 + lane) * 16);

        acc0 = __builtin_amdgcn_wmma_f32_16x16x32_f16(false, a, false, b0, (short)0, acc0, false, false);
        acc1 = __builtin_amdgcn_wmma_f32_16x16x32_f16(false, a, false, b1, (short)0, acc1, false, false);
        acc2 = __builtin_amdgcn_wmma_f32_16x16x32_f16(false, a, false, b2, (short)0, acc2, false, false);
    }

    // C/D layout: VGPR r -> M = r + 8*(lane>=16), N = lane&15
    const int n = lane & 15;
    const float bn0 = bias[n];
    const float bn1 = bias[16 + n];
    const float bn2 = (n < 4) ? bias[32 + n] : 0.f;
    const int mhi = (lane >> 4) << 3;
    for (int r = 0; r < 8; ++r) {
        int grow = mtile * 16 + mhi + r;
        float* orow = logits + (size_t)grow * NT;
        orow[n]      = lrelu(acc0[r] + bn0);
        orow[16 + n] = lrelu(acc1[r] + bn1);
        if (n < 4) orow[32 + n] = lrelu(acc2[r] + bn2);
    }
}

// ---------------------------------------------------------------------------
// Kernel 4: CRF forward (logsumexp scan) + gold score -> per-batch loss
// one block per batch element; 36 active tag lanes; trans & alpha in LDS.
// ---------------------------------------------------------------------------
__global__ __launch_bounds__(64) void crf_nll_kernel(
        const float* __restrict__ logits, const int* __restrict__ tags,
        const int* __restrict__ xlen, const float* __restrict__ trans,
        float* __restrict__ lossb) {
    __shared__ float tr[NT * NT];
    __shared__ float alpha[NT];
    __shared__ float red[64];
    const int b = blockIdx.x, tid = threadIdx.x;
    for (int i = tid; i < NT * NT; i += 64) tr[i] = trans[i];
    __syncthreads();

    const float* em = logits + (size_t)b * SS * NT;
    const int* tg = tags + b * SS;
    const int L = xlen[b];

    if (tid < NT) alpha[tid] = em[tid] + tr[TSTART * NT + tid];
    __syncthreads();

    for (int t = 1; t < SS; ++t) {
        float nv = 0.f;
        if (tid < NT) {
            float mx = NEG_INF;
            for (int i = 0; i < NT; ++i) {
                float v = alpha[i] + tr[i * NT + tid];
                mx = fmaxf(mx, v);
            }
            float sum = 0.f;
            for (int i = 0; i < NT; ++i) {
                float v = alpha[i] + tr[i * NT + tid];
                sum += expf(v - mx);
            }
            nv = mx + logf(sum) + em[t * NT + tid];
        }
        __syncthreads();
        if (tid < NT && t < L) alpha[tid] = nv;
        __syncthreads();
    }

    // gold: emission + transition scores (strided, ordered tree reduce)
    float part = 0.f;
    for (int t = tid; t < SS; t += 64) {
        if (t < L) part += em[t * NT + tg[t]];
        if (t >= 1 && t < L) part += tr[tg[t - 1] * NT + tg[t]];
    }
    red[tid] = part;
    __syncthreads();
    for (int s = 32; s > 0; s >>= 1) {
        if (tid < s) red[tid] += red[tid + s];
        __syncthreads();
    }

    if (tid == 0) {
        float mx = NEG_INF;
        for (int j = 0; j < NT; ++j) mx = fmaxf(mx, alpha[j] + tr[j * NT + TSTOP]);
        float sum = 0.f;
        for (int j = 0; j < NT; ++j) sum += expf(alpha[j] + tr[j * NT + TSTOP] - mx);
        float logZ = mx + logf(sum);
        int lastIdx = (L > 0) ? (L - 1) : 0;   // jnp take_along_axis clips
        int lastTag = tg[lastIdx];
        float gold = red[0] + tr[TSTART * NT + tg[0]] + tr[lastTag * NT + TSTOP];
        lossb[b] = logZ - gold;
    }
}

// ---------------------------------------------------------------------------
// Kernel 5: Viterbi (max-plus scan, backpointers in ws) + backtrace
// ---------------------------------------------------------------------------
__global__ __launch_bounds__(64) void crf_viterbi_kernel(
        const float* __restrict__ logits, const int* __restrict__ xlen,
        const float* __restrict__ trans, unsigned char* __restrict__ bp,
        float* __restrict__ out) {
    __shared__ float tr[NT * NT];
    __shared__ float delta[NT];
    __shared__ int   slast;
    __shared__ float sscore;
    const int b = blockIdx.x, tid = threadIdx.x;
    for (int i = tid; i < NT * NT; i += 64) tr[i] = trans[i];
    __syncthreads();

    const float* em = logits + (size_t)b * SS * NT;
    const int L = xlen[b];
    unsigned char* bpb = bp + (size_t)b * SS * NT;

    if (tid < NT) delta[tid] = em[tid] + tr[TSTART * NT + tid];
    __syncthreads();

    for (int t = 1; t < SS; ++t) {
        float best = NEG_INF;
        int bi = 0;
        if (tid < NT) {
            for (int i = 0; i < NT; ++i) {
                float v = delta[i] + tr[i * NT + tid];
                if (v > best) { best = v; bi = i; }   // first max on ties
            }
            best += em[t * NT + tid];
        }
        __syncthreads();
        if (tid < NT) {
            if (t < L) { delta[tid] = best; bpb[t * NT + tid] = (unsigned char)bi; }
            else       { bpb[t * NT + tid] = (unsigned char)tid; }  // identity
        }
        __syncthreads();
    }

    if (tid == 0) {
        float mx = NEG_INF; int am = 0;
        for (int j = 0; j < NT; ++j) {
            float v = delta[j] + tr[j * NT + TSTOP];
            if (v > mx) { mx = v; am = j; }
        }
        sscore = mx; slast = am;
    }
    __syncthreads();

    if (tid == 0) {
        float* path = out + 1 + (size_t)b * SS;
        int cur = slast;
        path[SS - 1] = ((SS - 1) < L) ? (float)cur : 0.f;
        for (int t = SS - 1; t >= 1; --t) {
            int prev = bpb[t * NT + cur];
            path[t - 1] = ((t - 1) < L) ? (float)prev : 0.f;
            cur = prev;
        }
        out[1 + BB * SS + b] = sscore;
    }
}

// ---------------------------------------------------------------------------
// Kernel 6: deterministic ordered sum of the 64 per-batch losses
// ---------------------------------------------------------------------------
__global__ __launch_bounds__(32) void sum_loss_kernel(const float* __restrict__ lossb,
                                                      float* __restrict__ out) {
    if (threadIdx.x == 0) {
        float s = 0.f;
        for (int i = 0; i < BB; ++i) s += lossb[i];
        out[0] = s;
    }
}

// ---------------------------------------------------------------------------
extern "C" void kernel_launch(void* const* d_in, const int* in_sizes, int n_in,
                              void* d_out, int out_size, void* d_ws, size_t ws_size,
                              hipStream_t stream) {
    const float* x1    = (const float*)d_in[0];   // (64,512,768)
    const int*   hidx  = (const int*)  d_in[1];   // (64,512)
    const int*   tags  = (const int*)  d_in[2];   // (64,512)
    const float* W     = (const float*)d_in[3];   // (768,36)
    const float* bias  = (const float*)d_in[4];   // (36,)
    const float* trans = (const float*)d_in[5];   // (36,36)
    float* out = (float*)d_out;                   // [loss | path (64*512) | score (64)]

    // workspace layout (256B aligned slabs)
    char* ws = (char*)d_ws;
    int*           xlen   = (int*)ws;                              // 256 B
    float*         lossb  = (float*)(ws + 512);                    // 256 B
    _Float16*      wfrag  = (_Float16*)(ws + 1024);                // 72*32*16*2 = 73728 B
    float*         logits = (float*)(ws + 81920);                  // 32768*36*4 = 4718592 B
    unsigned char* bp     = (unsigned char*)(ws + 81920 + 4718592);// 64*512*36  = 1179648 B

    xlen_kernel      <<<BB,  256, 0, stream>>>(hidx, xlen);
    packw_kernel     <<<72,   32, 0, stream>>>(W, wfrag);
    gemm_logits_kernel<<<256, 256, 0, stream>>>(x1, hidx, bias, wfrag, logits);
    crf_nll_kernel   <<<BB,   64, 0, stream>>>(logits, tags, xlen, trans, lossb);
    crf_viterbi_kernel<<<BB,  64, 0, stream>>>(logits, xlen, trans, bp, out);
    sum_loss_kernel  <<<1,    32, 0, stream>>>(lossb, out);
}
